// GridFMTransformer_65292092834136
// MI455X (gfx1250) — compile-verified, matrix operand order checked by reading, not run on hardware
//
#include <hip/hip_runtime.h>
#include <hip/hip_bf16.h>

// ---------------------------------------------------------------------------
// GridFM GPS transformer forward for MI455X (gfx1250, wave32, WMMA).
// GEMM: 128x64 block tile, 32x32 wave tile (4 WMMAs/k-step, 2x2 register
// blocking), tiles staged via global_load_async_to_lds_b128, fragments via
// ds_load_b128, math via v_wmma_f32_16x16x32_f16 (f32 acc).
// Attention keeps the full 64x1024 f32 score block in LDS (263KB < 320KB/WGP)
// and converts P to f16 in place for P@V.
// ---------------------------------------------------------------------------

typedef __attribute__((ext_vector_type(16))) _Float16 v16h;
typedef __attribute__((ext_vector_type(8)))  _Float16 v8h;
typedef __attribute__((ext_vector_type(4)))  _Float16 v4h;
typedef __attribute__((ext_vector_type(8)))  float    v8f;
typedef __attribute__((ext_vector_type(4)))  int      v4i;

#define NTOT   8192
#define CCH    256
#define NHEAD  8
#define DH     32
#define NG     8
#define NNODE  1024
#define NLAYER 8
#define NEDGE  32768
#define FFW    512
#define INDIM  9
#define PEDIM  20
#define ENCOUT 236
#define OUTDIM 6

#if __has_builtin(__builtin_amdgcn_global_load_async_to_lds_b128) && \
    __has_builtin(__builtin_amdgcn_s_wait_asynccnt)
#define USE_ASYNC_LDS 1
#else
#define USE_ASYNC_LDS 0
#endif

union Frag16 { v16h v; v8h h[2]; };

__device__ __forceinline__ v8f wmma16(const v16h& a, const v16h& b, const v8f& c) {
  return __builtin_amdgcn_wmma_f32_16x16x32_f16(false, a, false, b, (short)0, c,
                                                false, false);
}

__device__ __forceinline__ float lrelu(float v) { return v >= 0.0f ? v : 0.01f * v; }

__device__ __forceinline__ void stage16(const _Float16* src, _Float16* dstLds) {
#if USE_ASYNC_LDS
  __builtin_amdgcn_global_load_async_to_lds_b128((v4i*)src, (v4i*)dstLds, 0, 0);
#else
  *(v8h*)dstLds = *(const v8h*)src;
#endif
}

// ---------------------------------------------------------------------------
// WMMA GEMM (f16 A row-major, Bt = B transposed [N][K] -> f32 C):
// C[M,N] = act(A @ B + bias). Block 128x64, 8 waves of 32x32 (2x2 frags).
// ---------------------------------------------------------------------------
#define TM 128
#define TN 64
#define TK 32
#define APITCH 40  // halves; 80B rows keep 16B alignment for kh in {0,8,16,24}

__launch_bounds__(256)
__global__ void gemm16(const _Float16* __restrict__ A, const _Float16* __restrict__ Bt,
                       const float* __restrict__ bias, float* __restrict__ C,
                       int M, int N, int K, int act) {
  __shared__ _Float16 Ash[TM][APITCH];   // [m][k]
  __shared__ _Float16 BshT[TN][APITCH];  // [n][k]

  const int tid  = threadIdx.x;
  const int lane = tid & 31;
  const int wave = tid >> 5;
  const int wm   = wave & 3;   // 32-row block
  const int wn   = wave >> 2;  // 32-col block
  const int m0   = blockIdx.y * TM;
  const int n0   = blockIdx.x * TN;

  const int mrow = lane & 15;
  const int kh   = (lane >> 4) << 3;  // 0 or 8
  const int vh   = (lane >> 4) << 3;  // C-layout row offset

  // staging jobs: A = 512 b128 chunks (2/thread), B = 256 chunks (1/thread)
  const int ra0 = tid >> 2;                 // 0..63
  const int ra1 = (tid + 256) >> 2;         // 64..127
  const int seg = (tid & 3) << 3;           // 0,8,16,24

  v8f acc00 = {}, acc01 = {}, acc10 = {}, acc11 = {};

  for (int k0 = 0; k0 < K; k0 += TK) {
    stage16(A  + (size_t)(m0 + ra0) * K + k0 + seg, &Ash[ra0][seg]);
    stage16(A  + (size_t)(m0 + ra1) * K + k0 + seg, &Ash[ra1][seg]);
    stage16(Bt + (size_t)(n0 + ra0) * K + k0 + seg, &BshT[ra0][seg]);
    if (k0 + TK < K) {
      __builtin_prefetch(&A[(size_t)(m0 + ra0) * K + k0 + TK], 0, 1);
      __builtin_prefetch(&A[(size_t)(m0 + ra1) * K + k0 + TK], 0, 1);
      __builtin_prefetch(&Bt[(size_t)(n0 + ra0) * K + k0 + TK], 0, 1);
    }
#if USE_ASYNC_LDS
    __builtin_amdgcn_s_wait_asynccnt(0);
#endif
    __syncthreads();

    Frag16 a0, a1, b0, b1;
    a0.h[0] = *(const v8h*)&Ash[wm * 32 + mrow][kh];
    a0.h[1] = *(const v8h*)&Ash[wm * 32 + mrow][kh + 16];
    a1.h[0] = *(const v8h*)&Ash[wm * 32 + 16 + mrow][kh];
    a1.h[1] = *(const v8h*)&Ash[wm * 32 + 16 + mrow][kh + 16];
    b0.h[0] = *(const v8h*)&BshT[wn * 32 + mrow][kh];
    b0.h[1] = *(const v8h*)&BshT[wn * 32 + mrow][kh + 16];
    b1.h[0] = *(const v8h*)&BshT[wn * 32 + 16 + mrow][kh];
    b1.h[1] = *(const v8h*)&BshT[wn * 32 + 16 + mrow][kh + 16];

    acc00 = wmma16(a0.v, b0.v, acc00);
    acc01 = wmma16(a0.v, b1.v, acc01);
    acc10 = wmma16(a1.v, b0.v, acc10);
    acc11 = wmma16(a1.v, b1.v, acc11);
    __syncthreads();
  }

  const int r0 = m0 + wm * 32 + vh;        // rows of acc0x
  const int r1 = r0 + 16;                  // rows of acc1x
  const int c0 = n0 + wn * 32 + mrow;
  const int c1 = c0 + 16;
  const float bb0 = bias ? bias[c0] : 0.0f;
  const float bb1 = bias ? bias[c1] : 0.0f;
  #pragma unroll
  for (int v = 0; v < 8; ++v) {
    float x00 = acc00[v] + bb0, x01 = acc01[v] + bb1;
    float x10 = acc10[v] + bb0, x11 = acc11[v] + bb1;
    if (act == 1) {
      x00 = fmaxf(x00, 0.f); x01 = fmaxf(x01, 0.f);
      x10 = fmaxf(x10, 0.f); x11 = fmaxf(x11, 0.f);
    } else if (act == 2) {
      x00 = lrelu(x00); x01 = lrelu(x01); x10 = lrelu(x10); x11 = lrelu(x11);
    }
    C[(size_t)(r0 + v) * N + c0] = x00;
    C[(size_t)(r0 + v) * N + c1] = x01;
    C[(size_t)(r1 + v) * N + c0] = x10;
    C[(size_t)(r1 + v) * N + c1] = x11;
  }
}

// ---------------------------------------------------------------------------
// Attention. qkv16: f16 [NTOT, 768] (Q|K|V). vt16: f16 [g][h][d][key].
// One block per (g, head, 64-query block). S block 64x1028 f32 in LDS.
// ---------------------------------------------------------------------------
#define SPITCH 1028  // floats; 4112B rows: 16B-aligned, 4-bank row skew

__launch_bounds__(256)
__global__ void attn_kernel(const _Float16* __restrict__ qkv16,
                            const _Float16* __restrict__ vt16,
                            float* __restrict__ attout) {
  extern __shared__ float Sl[];  // [64][SPITCH]

  const int bid  = blockIdx.x;
  const int qb   = bid & 15;
  const int hh   = (bid >> 4) & 7;
  const int g    = bid >> 7;
  const int tid  = threadIdx.x;
  const int lane = tid & 31;
  const int wave = tid >> 5;
  const int mrow = lane & 15;
  const int kh   = (lane >> 4) << 3;
  const int vh   = (lane >> 4) << 3;

  const _Float16* Qb = qkv16 + (size_t)(g * NNODE + qb * 64) * (3 * CCH) + hh * DH;
  const _Float16* Kb = qkv16 + (size_t)(g * NNODE) * (3 * CCH) + CCH + hh * DH;
  const _Float16* Vt = vt16 + (size_t)(g * NHEAD + hh) * DH * NNODE;
  const float scale = 0.17677669529663687f;  // 1/sqrt(32)

  // ---- S(64x1024) = Q(64x32) @ K^T(32x1024); K-dim == dh == 32 ----
  {
    const int rt = wave & 3;
    Frag16 qf;
    qf.h[0] = *(const v8h*)&Qb[(size_t)(rt * 16 + mrow) * (3 * CCH) + kh];
    qf.h[1] = *(const v8h*)&Qb[(size_t)(rt * 16 + mrow) * (3 * CCH) + kh + 16];
    for (int ct = (wave >> 2); ct < 64; ct += 2) {
      const int key = ct * 16 + mrow;
      Frag16 kf;
      kf.h[0] = *(const v8h*)&Kb[(size_t)key * (3 * CCH) + kh];
      kf.h[1] = *(const v8h*)&Kb[(size_t)key * (3 * CCH) + kh + 16];
      v8f s = {};
      s = wmma16(qf.v, kf.v, s);
      #pragma unroll
      for (int v = 0; v < 8; ++v)
        Sl[(rt * 16 + vh + v) * SPITCH + ct * 16 + mrow] = s[v] * scale;
    }
  }
  __syncthreads();

  // ---- softmax per row; write P back in place as f16 (front of the row) ----
  if (tid < 64) {
    float* row = &Sl[tid * SPITCH];
    _Float16* p16 = (_Float16*)row;
    float mx = -1e30f;
    for (int j = 0; j < NNODE; ++j) mx = fmaxf(mx, row[j]);
    float sum = 0.f;
    for (int j = 0; j < NNODE; ++j) sum += __expf(row[j] - mx);
    const float inv = 1.0f / sum;
    for (int j = 0; j < NNODE; ++j) {
      float p = __expf(row[j] - mx) * inv;   // read f32 before overwriting bytes
      p16[j] = (_Float16)p;                  // j/2 <= j: safe in-place narrow
    }
  }
  __syncthreads();

  // ---- O(64x32) = P(64x1024) @ V(1024x32): 4 row-tiles x 2 col-tiles ----
  {
    const int rt = wave & 3;
    const int nt = wave >> 2;
    const _Float16* p16 = (const _Float16*)&Sl[(size_t)(rt * 16 + mrow) * SPITCH];
    const _Float16* vcol = &Vt[(size_t)(nt * 16 + mrow) * NNODE];
    v8f o = {};
    for (int ks = 0; ks < 32; ++ks) {
      Frag16 pf, vf;
      pf.h[0] = *(const v8h*)&p16[ks * 32 + kh];
      pf.h[1] = *(const v8h*)&p16[ks * 32 + kh + 16];
      vf.h[0] = *(const v8h*)&vcol[ks * 32 + kh];
      vf.h[1] = *(const v8h*)&vcol[ks * 32 + kh + 16];
      o = wmma16(pf.v, vf.v, o);
    }
    #pragma unroll
    for (int v = 0; v < 8; ++v) {
      int q = qb * 64 + rt * 16 + vh + v;
      attout[(size_t)(g * NNODE + q) * CCH + hh * DH + nt * 16 + mrow] = o[v];
    }
  }
}

// ---------------------------------------------------------------------------
// Streaming f32 -> f16 conversion (n % 4 == 0)
// ---------------------------------------------------------------------------
__global__ void cvt16(const float* __restrict__ in, _Float16* __restrict__ out, int n) {
  for (int i = (blockIdx.x * blockDim.x + threadIdx.x) * 4; i < n;
       i += gridDim.x * blockDim.x * 4) {
    float4 v = *(const float4*)(in + i);
    v4h o;
    o[0] = (_Float16)v.x; o[1] = (_Float16)v.y;
    o[2] = (_Float16)v.z; o[3] = (_Float16)v.w;
    *(v4h*)(out + i) = o;
  }
}

// Layered weight transpose-convert: out[l][n][k] = (f16) in[l][k][n]
__global__ void cvtT(const float* __restrict__ in, _Float16* __restrict__ out,
                     int K, int N, int L) {
  const int per = K * N;
  const int total = per * L;
  for (int idx = blockIdx.x * blockDim.x + threadIdx.x; idx < total;
       idx += gridDim.x * blockDim.x) {
    const int l = idx / per;
    const int rem = idx - l * per;
    const int n = rem / K;
    const int k = rem - n * K;
    out[idx] = (_Float16)in[(size_t)l * per + (size_t)k * N + n];
  }
}

// qkv f32 -> V transposed f16: vt16[g][h][d][key] = qkv[(g*N+key)*768 + 512 + h*32 + d]
__global__ void vtrans(const float* __restrict__ qkv, _Float16* __restrict__ vt16) {
  const int total = NG * NHEAD * DH * NNODE;
  for (int idx = blockIdx.x * blockDim.x + threadIdx.x; idx < total;
       idx += gridDim.x * blockDim.x) {
    const int key = idx & (NNODE - 1);
    const int d   = (idx >> 10) & (DH - 1);
    const int hh  = (idx >> 15) & (NHEAD - 1);
    const int g   = idx >> 18;
    vt16[idx] = (_Float16)qkv[(size_t)(g * NNODE + key) * (3 * CCH) + 2 * CCH + hh * DH + d];
  }
}

// ---------------------------------------------------------------------------
// GINE edge messages: msg = relu(h[src] + ea@eW + eb) scatter-added into agg
// ---------------------------------------------------------------------------
__global__ void gine_edge(const float* __restrict__ h, const float* __restrict__ ea,
                          const int* __restrict__ src, const int* __restrict__ dst,
                          const float* __restrict__ eW, const float* __restrict__ eb,
                          float* __restrict__ agg) {
  const int total = NEDGE * CCH;
  for (int idx = blockIdx.x * blockDim.x + threadIdx.x; idx < total;
       idx += gridDim.x * blockDim.x) {
    const int e = idx >> 8;
    const int c = idx & 255;
    float lin = ea[e * 2] * eW[c] + ea[e * 2 + 1] * eW[CCH + c] + eb[c];
    float m = fmaxf(h[(size_t)src[e] * CCH + c] + lin, 0.0f);
    atomicAdd(&agg[(size_t)dst[e] * CCH + c], m);
  }
}

// ---------------------------------------------------------------------------
// Training-mode BatchNorm over rows (biased var), optional residual add.
// ---------------------------------------------------------------------------
__launch_bounds__(256)
__global__ void bn_kernel(const float* __restrict__ X, const float* __restrict__ R,
                          float* __restrict__ Y, const float* __restrict__ g,
                          const float* __restrict__ b, int rows, int ldx, int ldr, int ldy) {
  __shared__ float s1[256], s2[256];
  const int c = blockIdx.x;
  const int tid = threadIdx.x;
  float sum = 0.f, sq = 0.f;
  for (int i = tid; i < rows; i += 256) {
    float v = X[(size_t)i * ldx + c];
    if (R) v += R[(size_t)i * ldr + c];
    sum += v; sq += v * v;
  }
  s1[tid] = sum; s2[tid] = sq;
  __syncthreads();
  for (int s = 128; s > 0; s >>= 1) {
    if (tid < s) { s1[tid] += s1[tid + s]; s2[tid] += s2[tid + s]; }
    __syncthreads();
  }
  const float mean = s1[0] / rows;
  const float var  = s2[0] / rows - mean * mean;
  const float rs   = rsqrtf(var + 1e-5f);
  const float gg = g[c], bb = b[c];
  for (int i = tid; i < rows; i += 256) {
    float v = X[(size_t)i * ldx + c];
    if (R) v += R[(size_t)i * ldr + c];
    Y[(size_t)i * ldy + c] = (v - mean) * rs * gg + bb;
  }
}

// ---------------------------------------------------------------------------
// Small helpers
// ---------------------------------------------------------------------------
__global__ void zero_kernel(float* __restrict__ p, int n) {
  for (int i = blockIdx.x * blockDim.x + threadIdx.x; i < n; i += gridDim.x * blockDim.x)
    p[i] = 0.0f;
}

__global__ void add_kernel(const float* __restrict__ a, const float* __restrict__ b,
                           float* __restrict__ o, int n) {
  for (int i = blockIdx.x * blockDim.x + threadIdx.x; i < n; i += gridDim.x * blockDim.x)
    o[i] = a[i] + b[i];
}

__global__ void enc_kernel(const float* __restrict__ x, const float* __restrict__ W,
                           const float* __restrict__ b, float* __restrict__ out) {
  const int total = NTOT * ENCOUT;
  for (int idx = blockIdx.x * blockDim.x + threadIdx.x; idx < total;
       idx += gridDim.x * blockDim.x) {
    const int r = idx / ENCOUT, c = idx % ENCOUT;
    float s = b[c];
    #pragma unroll
    for (int k = 0; k < INDIM; ++k) s += x[r * INDIM + k] * W[k * ENCOUT + c];
    out[idx] = lrelu(s);
  }
}

__global__ void dec_kernel(const float* __restrict__ t, const float* __restrict__ W,
                           const float* __restrict__ b, float* __restrict__ out) {
  const int total = NTOT * OUTDIM;
  for (int idx = blockIdx.x * blockDim.x + threadIdx.x; idx < total;
       idx += gridDim.x * blockDim.x) {
    const int r = idx / OUTDIM, c = idx % OUTDIM;
    float s = b[c];
    for (int k = 0; k < CCH; ++k) s += t[(size_t)r * CCH + k] * W[k * OUTDIM + c];
    out[idx] = s;
  }
}

// ---------------------------------------------------------------------------
// Host launcher
// ---------------------------------------------------------------------------
extern "C" void kernel_launch(void* const* d_in, const int* in_sizes, int n_in,
                              void* d_out, int out_size, void* d_ws, size_t ws_size,
                              hipStream_t stream) {
  const float* x     = (const float*)d_in[0];
  const float* pe    = (const float*)d_in[1];
  const float* ea    = (const float*)d_in[2];
  const int*   ei    = (const int*)d_in[3];
  const float* encW  = (const float*)d_in[4];
  const float* encb  = (const float*)d_in[5];
  const float* in_g  = (const float*)d_in[6];
  const float* in_b  = (const float*)d_in[7];
  const float* pe_g  = (const float*)d_in[8];
  const float* pe_b  = (const float*)d_in[9];
  const float* geW   = (const float*)d_in[10];
  const float* geb   = (const float*)d_in[11];
  const float* gmW   = (const float*)d_in[12];
  const float* gmb   = (const float*)d_in[13];
  const float* aiW   = (const float*)d_in[14];
  const float* aib   = (const float*)d_in[15];
  const float* aoW   = (const float*)d_in[16];
  const float* aob   = (const float*)d_in[17];
  const float* n1g   = (const float*)d_in[18];
  const float* n1b   = (const float*)d_in[19];
  const float* n2g   = (const float*)d_in[20];
  const float* n2b   = (const float*)d_in[21];
  const float* n3g   = (const float*)d_in[22];
  const float* n3b   = (const float*)d_in[23];
  const float* fW1   = (const float*)d_in[24];
  const float* fb1   = (const float*)d_in[25];
  const float* fW2   = (const float*)d_in[26];
  const float* fb2   = (const float*)d_in[27];
  const float* lng   = (const float*)d_in[28];
  const float* lnb   = (const float*)d_in[29];
  const float* pre_g = (const float*)d_in[30];
  const float* pre_b = (const float*)d_in[31];
  const float* dW1   = (const float*)d_in[32];
  const float* db1   = (const float*)d_in[33];
  const float* dW2   = (const float*)d_in[34];
  const float* db2   = (const float*)d_in[35];

  const int NC = NTOT * CCH;  // 2,097,152
  float* h    = (float*)d_ws;
  float* t1   = h  + NC;
  float* t2   = t1 + NC;
  float* t3   = t2 + NC;
  float* qkvb = t3 + NC;                // NTOT*768 f32
  float* mid  = qkvb + NTOT * 3 * CCH;  // NTOT*512 f32

  _Float16* a16    = (_Float16*)(mid + NTOT * FFW);  // max NTOT*512 halves
  _Float16* qkv16  = a16 + NTOT * FFW;               // NTOT*768 halves
  _Float16* vt16   = qkv16 + NTOT * 3 * CCH;         // G*H*32*1024 halves
  _Float16* gmWt   = vt16 + NG * NHEAD * DH * NNODE; // transposed f16 weights
  _Float16* aiWt   = gmWt + NLAYER * CCH * CCH;
  _Float16* aoWt   = aiWt + NLAYER * CCH * 3 * CCH;
  _Float16* fW1t   = aoWt + NLAYER * CCH * CCH;
  _Float16* fW2t   = fW1t + NLAYER * CCH * FFW;
  _Float16* dW1t   = fW2t + NLAYER * FFW * CCH;

  const int attn_lds = 64 * SPITCH * 4;  // 263168B dynamic LDS (< 320KB WGP)
  (void)hipFuncSetAttribute((const void*)attn_kernel,
                            hipFuncAttributeMaxDynamicSharedMemorySize, attn_lds);

  const dim3 blk(256);
  const dim3 gElem((NC + 255) / 256);
  const int* src = ei;
  const int* dst = ei + NEDGE;

  auto cvtGrid = [](int n) { return dim3((n / 4 + 255) / 256); };
  auto tGrid   = [](int n) { return dim3((n + 255) / 256); };

  // ---- weight transpose+convert (deterministic, every call) ----
  cvtT<<<tGrid(NLAYER * CCH * CCH), blk, 0, stream>>>(gmW, gmWt, CCH, CCH, NLAYER);
  cvtT<<<tGrid(NLAYER * CCH * 3 * CCH), blk, 0, stream>>>(aiW, aiWt, CCH, 3 * CCH, NLAYER);
  cvtT<<<tGrid(NLAYER * CCH * CCH), blk, 0, stream>>>(aoW, aoWt, CCH, CCH, NLAYER);
  cvtT<<<tGrid(NLAYER * CCH * FFW), blk, 0, stream>>>(fW1, fW1t, CCH, FFW, NLAYER);
  cvtT<<<tGrid(NLAYER * FFW * CCH), blk, 0, stream>>>(fW2, fW2t, FFW, CCH, NLAYER);
  cvtT<<<tGrid(CCH * CCH), blk, 0, stream>>>(dW1, dW1t, CCH, CCH, 1);

  // ---- encoder: h = [BN(lrelu(x@encW+encb)) | BN(pe)] ----
  enc_kernel<<<dim3((NTOT * ENCOUT + 255) / 256), blk, 0, stream>>>(x, encW, encb, t1);
  bn_kernel<<<dim3(ENCOUT), blk, 0, stream>>>(t1, nullptr, h, in_g, in_b,
                                              NTOT, ENCOUT, 0, CCH);
  bn_kernel<<<dim3(PEDIM), blk, 0, stream>>>(pe, nullptr, h + ENCOUT, pe_g, pe_b,
                                             NTOT, PEDIM, 0, CCH);

  for (int l = 0; l < NLAYER; ++l) {
    // ---- local GINE branch ----
    zero_kernel<<<gElem, blk, 0, stream>>>(t2, NC);
    gine_edge<<<dim3((NEDGE * CCH + 255) / 256), blk, 0, stream>>>(
        h, ea, src, dst, geW + l * 2 * CCH, geb + l * CCH, t2);
    add_kernel<<<gElem, blk, 0, stream>>>(h, t2, t3, NC);
    cvt16<<<cvtGrid(NC), blk, 0, stream>>>(t3, a16, NC);
    gemm16<<<dim3(CCH / TN, NTOT / TM), blk, 0, stream>>>(
        a16, gmWt + l * CCH * CCH, gmb + l * CCH, t1, NTOT, CCH, CCH, 2);
    bn_kernel<<<dim3(CCH), blk, 0, stream>>>(t1, h, t1, n1g + l * CCH, n1b + l * CCH,
                                             NTOT, CCH, CCH, CCH);

    // ---- global attention branch ----
    cvt16<<<cvtGrid(NC), blk, 0, stream>>>(h, a16, NC);
    gemm16<<<dim3(3 * CCH / TN, NTOT / TM), blk, 0, stream>>>(
        a16, aiWt + l * CCH * 3 * CCH, aib + l * 3 * CCH, qkvb, NTOT, 3 * CCH, CCH, 0);
    cvt16<<<cvtGrid(NTOT * 3 * CCH), blk, 0, stream>>>(qkvb, qkv16, NTOT * 3 * CCH);
    vtrans<<<dim3((NG * NHEAD * DH * NNODE + 255) / 256), blk, 0, stream>>>(qkvb, vt16);
    attn_kernel<<<dim3(NG * NHEAD * (NNODE / 64)), blk, attn_lds, stream>>>(qkv16, vt16, t2);
    cvt16<<<cvtGrid(NC), blk, 0, stream>>>(t2, a16, NC);
    gemm16<<<dim3(CCH / TN, NTOT / TM), blk, 0, stream>>>(
        a16, aoWt + l * CCH * CCH, aob + l * CCH, t3, NTOT, CCH, CCH, 0);
    bn_kernel<<<dim3(CCH), blk, 0, stream>>>(t3, h, t3, n2g + l * CCH, n2b + l * CCH,
                                             NTOT, CCH, CCH, CCH);

    // ---- combine + FFN ----
    add_kernel<<<gElem, blk, 0, stream>>>(t1, t3, t2, NC);
    cvt16<<<cvtGrid(NC), blk, 0, stream>>>(t2, a16, NC);
    gemm16<<<dim3(FFW / TN, NTOT / TM), blk, 0, stream>>>(
        a16, fW1t + l * CCH * FFW, fb1 + l * FFW, mid, NTOT, FFW, CCH, 1);
    cvt16<<<cvtGrid(NTOT * FFW), blk, 0, stream>>>(mid, a16, NTOT * FFW);
    gemm16<<<dim3(CCH / TN, NTOT / TM), blk, 0, stream>>>(
        a16, fW2t + l * FFW * CCH, fb2 + l * CCH, t1, NTOT, CCH, FFW, 0);
    bn_kernel<<<dim3(CCH), blk, 0, stream>>>(t1, t2, t1, n3g + l * CCH, n3b + l * CCH,
                                             NTOT, CCH, CCH, CCH);
    bn_kernel<<<dim3(CCH), blk, 0, stream>>>(t1, nullptr, h, lng + l * CCH, lnb + l * CCH,
                                             NTOT, CCH, CCH, CCH);
  }

  // ---- decoder ----
  bn_kernel<<<dim3(CCH), blk, 0, stream>>>(h, nullptr, t1, pre_g, pre_b,
                                           NTOT, CCH, CCH, CCH);
  cvt16<<<cvtGrid(NC), blk, 0, stream>>>(t1, a16, NC);
  gemm16<<<dim3(CCH / TN, NTOT / TM), blk, 0, stream>>>(
      a16, dW1t, db1, t2, NTOT, CCH, CCH, 2);
  dec_kernel<<<dim3((NTOT * OUTDIM + 255) / 256), blk, 0, stream>>>(
      t2, dW2, db2, (float*)d_out);
}
